// SimpleResidualVQ_21483426415070
// MI455X (gfx1250) — matrix operand order repaired and checked by reading, not compiled, not run on hardware
//
#include <hip/hip_runtime.h>
#include <hip/hip_bf16.h>

typedef __attribute__((ext_vector_type(16))) _Float16 v16h;
typedef __attribute__((ext_vector_type(8)))  _Float16 v8h;
typedef __attribute__((ext_vector_type(8)))  float    v8f;

#define B_    16
#define F_    1024
#define DIN_  768
#define NCB_  8
#define K_    1024
#define D_    128
#define N_    (B_ * F_)          // 16384 rows
#define COMMIT_ 0.25f

// ---------------------------------------------------------------------------
// Fragment loaders.
// A (16x32, MxK) f16 layout: lanes 0-15 row=l, elems = K{k0..k0+7, k0+16..k0+23};
//                            lanes 16-31 shift the K window by +8.
// B (32x16, KxN) f16 layout: lanes 0-15 col=l, elems = K{k0..k0+7, k0+16..k0+23};
//                            lanes 16-31 shift the K window by +8.
// ---------------------------------------------------------------------------
__device__ inline v16h load_a_frag_f32(const float* rowptr, int k0, int half) {
    v16h v;
    int c0 = k0 + half * 8;
#pragma unroll
    for (int j = 0; j < 8; ++j) {
        v[j]     = (_Float16)rowptr[c0 + j];
        v[8 + j] = (_Float16)rowptr[c0 + 16 + j];
    }
    return v;
}

__device__ inline v16h load_a_frag_h(const _Float16* rowptr, int k0, int half) {
    int c0 = k0 + half * 8;
    v8h lo = *(const v8h*)(rowptr + c0);        // 16B aligned
    v8h hi = *(const v8h*)(rowptr + c0 + 16);   // 16B aligned
    v16h v;
#pragma unroll
    for (int j = 0; j < 8; ++j) { v[j] = lo[j]; v[8 + j] = hi[j]; }
    return v;
}

__device__ inline v16h load_b_frag_h(const _Float16* colptr, int k0, int half) {
    int c0 = k0 + half * 8;
    v8h lo = *(const v8h*)(colptr + c0);        // 16B aligned
    v8h hi = *(const v8h*)(colptr + c0 + 16);   // 16B aligned
    v16h v;
#pragma unroll
    for (int j = 0; j < 8; ++j) { v[j] = lo[j]; v[8 + j] = hi[j]; }
    return v;
}

// ---------------------------------------------------------------------------
// Wp (DIN x D) fp32 -> WpTh (D x DIN) f16: contiguous B-columns, pre-converted.
// ---------------------------------------------------------------------------
__global__ void k_prep_wp(const float* __restrict__ Wp, _Float16* __restrict__ WpTh) {
    int t = blockIdx.x * blockDim.x + threadIdx.x;     // t = c*DIN_ + r
    if (t >= DIN_ * D_) return;
    int c = t / DIN_;
    int r = t % DIN_;
    WpTh[t] = (_Float16)Wp[r * D_ + c];
}

// ---------------------------------------------------------------------------
// Per codebook row: ||e||^2 (fp32) and an f16 copy of the row.
// Thread 0 zeroes the loss accumulator.
// ---------------------------------------------------------------------------
__global__ void k_prep_e(const float* __restrict__ E, _Float16* __restrict__ Eh,
                         float* __restrict__ normE, float* __restrict__ lossAcc) {
    int t = blockIdx.x * blockDim.x + threadIdx.x;
    if (t == 0) *lossAcc = 0.0f;
    if (t >= NCB_ * K_) return;
    const float* row = E + (size_t)t * D_;
    _Float16*    hrw = Eh + (size_t)t * D_;
    float s = 0.0f;
#pragma unroll 4
    for (int j = 0; j < D_; ++j) {
        float e = row[j];
        s += e * e;
        hrw[j] = (_Float16)e;
    }
    normE[t] = s;
}

// ---------------------------------------------------------------------------
// Projection: R = z @ Wp + bp  (fp32 result + f16 shadow Rh).
// 16 rows per block, 8 waves x one 16-col tile, 24 WMMA K-steps over DIN.
// ---------------------------------------------------------------------------
__global__ __launch_bounds__(256)
void k_project(const float* __restrict__ z, const _Float16* __restrict__ WpTh,
               const float* __restrict__ bp, float* __restrict__ R,
               _Float16* __restrict__ Rh) {
    int r0   = blockIdx.x * 16;
    int wave = threadIdx.x >> 5;
    int lane = threadIdx.x & 31;
    int half = lane >> 4;
    int lr   = lane & 15;
    int n0   = wave * 16;                       // column tile in D

    const float*    arow = z    + (size_t)(r0 + lr) * DIN_;
    const _Float16* bcol = WpTh + (size_t)(n0 + lr) * DIN_;

    v8f c = {};
#pragma unroll 4
    for (int k0 = 0; k0 < DIN_; k0 += 32) {
        v16h a = load_a_frag_f32(arow, k0, half);
        v16h b = load_b_frag_h(bcol, k0, half);
        c = __builtin_amdgcn_wmma_f32_16x16x32_f16(false, a, false, b,
                                                   (short)0, c, false, false);
    }
    int col = n0 + lr;
    float bias = bp[col];
#pragma unroll
    for (int r = 0; r < 8; ++r) {
        int row = r0 + r + half * 8;
        float val = c[r] + bias;
        size_t off = (size_t)row * D_ + col;
        R[off]  = val;
        Rh[off] = (_Float16)val;
    }
}

// ---------------------------------------------------------------------------
// One codebook step: scores = ||e||^2 - 2 R.E^T (pure-f16 WMMA feeds),
// per-row argmin, residual update (fp32 + f16 shadow), loss accumulation.
// Block = 16 rows; 8 waves, each covers 8 column-tiles of 16 (K = 1024).
// ---------------------------------------------------------------------------
__global__ __launch_bounds__(256)
void k_vq_step(float* __restrict__ R, _Float16* __restrict__ Rh,
               const _Float16* __restrict__ Eh, const float* __restrict__ Ecb,
               const float* __restrict__ normEcb, float* __restrict__ codes_out,
               float* __restrict__ lossAcc, int cb) {
    __shared__ float sVal[8][16];
    __shared__ int   sIdx[8][16];
    __shared__ int   sBest[16];
    __shared__ float sLoss[8];

    int r0   = blockIdx.x * 16;
    int wave = threadIdx.x >> 5;
    int lane = threadIdx.x & 31;
    int half = lane >> 4;
    int lr   = lane & 15;

    const _Float16* arow = Rh + (size_t)(r0 + lr) * D_;

    // Preload the A (residual) fragments for all 4 K-steps (D=128).
    v16h a[4];
#pragma unroll
    for (int ks = 0; ks < 4; ++ks) a[ks] = load_a_frag_h(arow, ks * 32, half);

    float bestV[8];
    int   bestI[8];
#pragma unroll
    for (int r = 0; r < 8; ++r) { bestV[r] = 3.402823466e38f; bestI[r] = 0; }

    for (int ct = 0; ct < 8; ++ct) {
        int n0 = (wave * 8 + ct) * 16;
        const _Float16* bcol = Eh + (size_t)(n0 + lr) * D_;
        v8f c = {};
#pragma unroll
        for (int ks = 0; ks < 4; ++ks) {
            v16h b = load_b_frag_h(bcol, ks * 32, half);
            c = __builtin_amdgcn_wmma_f32_16x16x32_f16(false, a[ks], false, b,
                                                       (short)0, c, false, false);
        }
        int   n  = n0 + lr;
        float nE = normEcb[n];
#pragma unroll
        for (int r = 0; r < 8; ++r) {
            float v  = nE - 2.0f * c[r];   // ||x||^2 constant per row: drop it
            int  idx = n;
#pragma unroll
            for (int off = 8; off >= 1; off >>= 1) {   // width-16 half-wave min
                float ov = __shfl_xor(v, off, 16);
                int   oi = __shfl_xor(idx, off, 16);
                if (ov < v || (ov == v && oi < idx)) { v = ov; idx = oi; }
            }
            if (v < bestV[r] || (v == bestV[r] && idx < bestI[r])) {
                bestV[r] = v; bestI[r] = idx;
            }
        }
    }

    // Wave-local per-row winners -> LDS (row = r + half*8).
    if (lr == 0) {
#pragma unroll
        for (int r = 0; r < 8; ++r) {
            sVal[wave][r + half * 8] = bestV[r];
            sIdx[wave][r + half * 8] = bestI[r];
        }
    }
    __syncthreads();

    // Combine the 8 waves (ascending column ranges -> first-min tie-break).
    if (threadIdx.x < 16) {
        int row = threadIdx.x;
        float v  = sVal[0][row];
        int  idx = sIdx[0][row];
#pragma unroll
        for (int w = 1; w < 8; ++w) {
            float ov = sVal[w][row];
            int   oi = sIdx[w][row];
            if (ov < v || (ov == v && oi < idx)) { v = ov; idx = oi; }
        }
        sBest[row] = idx;
        codes_out[(size_t)(r0 + row) * NCB_ + cb] = (float)idx;
    }
    __syncthreads();

    // Residual update + commitment loss: 16x128 elements, 8 per thread.
    // Exact arithmetic in fp32; also refresh the f16 shadow for the next step.
    float lsum = 0.0f;
#pragma unroll
    for (int e = 0; e < 8; ++e) {
        int elem = threadIdx.x + e * 256;
        int row  = elem >> 7;
        int col  = elem & 127;
        int idx  = sBest[row];
        float ev = Ecb[(size_t)idx * D_ + col];
        size_t off = (size_t)(r0 + row) * D_ + col;
        float rv = R[off];
        float d  = ev - rv;          // stop_grad(z_q) - residual
        lsum += d * d;
        float nr = rv - ev;
        R[off]  = nr;
        Rh[off] = (_Float16)nr;
    }
#pragma unroll
    for (int off = 16; off >= 1; off >>= 1) lsum += __shfl_xor(lsum, off, 32);
    if (lane == 0) sLoss[wave] = lsum;
    __syncthreads();
    if (threadIdx.x == 0) {
        float s = 0.0f;
#pragma unroll
        for (int w = 0; w < 8; ++w) s += sLoss[w];
        atomicAdd(lossAcc, s);
    }
}

__global__ void k_finalize(const float* __restrict__ lossAcc, float* __restrict__ out) {
    out[0] = COMMIT_ * (*lossAcc) / (float)((size_t)N_ * D_);
}

// ---------------------------------------------------------------------------
extern "C" void kernel_launch(void* const* d_in, const int* in_sizes, int n_in,
                              void* d_out, int out_size, void* d_ws, size_t ws_size,
                              hipStream_t stream) {
    const float* z  = (const float*)d_in[0];   // (16,1024,768)
    const float* Wp = (const float*)d_in[1];   // (768,128)
    const float* bp = (const float*)d_in[2];   // (128,)
    const float* E  = (const float*)d_in[3];   // (8,1024,128)
    float* out = (float*)d_out;                // codes (16*1024*8) then loss (1)

    // Workspace layout (all partitions 256B-aligned by construction).
    char* ws = (char*)d_ws;
    float*    R       = (float*)ws;                            // N*D fp32   (8 MB)
    _Float16* Rh      = (_Float16*)(ws + (size_t)N_ * D_ * 4); // N*D f16    (4 MB)
    _Float16* Eh      = Rh + (size_t)N_ * D_;                  // NCB*K*D f16 (2 MB)
    _Float16* WpTh    = Eh + (size_t)NCB_ * K_ * D_;           // D*DIN f16  (192 KB)
    float*    normE   = (float*)(WpTh + (size_t)D_ * DIN_);    // NCB*K fp32 (32 KB)
    float*    lossAcc = normE + (size_t)NCB_ * K_;             // 1 float

    k_prep_wp<<<(DIN_ * D_ + 255) / 256, 256, 0, stream>>>(Wp, WpTh);
    k_prep_e<<<(NCB_ * K_ + 255) / 256, 256, 0, stream>>>(E, Eh, normE, lossAcc);
    k_project<<<N_ / 16, 256, 0, stream>>>(z, WpTh, bp, R, Rh);

    for (int cb = 0; cb < NCB_; ++cb) {
        k_vq_step<<<N_ / 16, 256, 0, stream>>>(R, Rh,
                                               Eh + (size_t)cb * K_ * D_,
                                               E + (size_t)cb * K_ * D_,
                                               normE + (size_t)cb * K_,
                                               out, lossAcc, cb);
    }
    k_finalize<<<1, 1, 0, stream>>>(lossAcc, out + (size_t)N_ * NCB_);
}
